// DownSample_1821066133669
// MI455X (gfx1250) — matrix operand order repaired and checked by reading, not compiled, not run on hardware
//
#include <hip/hip_runtime.h>
#include <hip/hip_bf16.h>

// ---------------------------------------------------------------------------
// CDNA5 (gfx1250) implementation of the DownSample block.
//
// Pipeline (all on `stream`, sequentially dependent):
//   1. fps_knn_kernel      : FPS (m=16) + 2-NN per center, writes coor_s out.
//   2. gemm_bf16_wmma<Sp>  : C[512,4096]  = sp_w x gathered sparse features.
//   3. channel_stats       : per-channel mean/var (batch-stat BN).
//   4. sp_final            : BN + LeakyReLU + max over k=2 -> d_out[0:1M).
//   5. gemm_bf16_wmma<Dn>  : C[256,131072] = dn_w x gathered dense features.
//   6. channel_stats, dn_h : BN + act + max over k=2 -> H [b,256,16,32].
//   7. gemm_bf16_wmma<Ds>  : C[256,32768] = ds_w(im2col k=3,s=2,p=1) x H.
//   8. channel_stats, ds_final -> d_out[1M:9.4M).
//
// Precision: bf16 A/B fragments, f32 WMMA accumulation
// (v_wmma_f32_16x16x32_bf16). Conv biases cancel inside batch-stat BN and
// are dropped (exact).
// Workspace: ~243 MB carved from d_ws (raw GEMM outputs kept for BN stats).
// ---------------------------------------------------------------------------

typedef __attribute__((ext_vector_type(16))) __bf16 v16bf;
typedef __attribute__((ext_vector_type(8)))  __bf16 v8bf;
typedef __attribute__((ext_vector_type(8)))  float  v8f;

#define BS      128
#define NSTK    32
#define MCTR    16      // n_stk_center
#define KNN_K   2
#define SP_IN   512
#define SP_OUT  512
#define DN_IN   256
#define DN_OUT  256
#define NPNT    32
#define EPSBN   1e-5f
#define SLOPE   0.2f

// ------------------------------ FPS + KNN ----------------------------------
// One wave32 per batch; lane == point index.
__global__ __launch_bounds__(32) void fps_knn_kernel(
    const float* __restrict__ coor, int* __restrict__ fps,
    int* __restrict__ knn, float* __restrict__ coor_out)
{
  const int b = blockIdx.x;
  const int lane = threadIdx.x;
  const float* cb = coor + (size_t)b * NSTK * 3;
  float x = cb[lane * 3 + 0];
  float y = cb[lane * 3 + 1];
  float z = cb[lane * 3 + 2];

  __shared__ int sidx[MCTR];

  // ---- farthest point sampling (record-then-update, start at 0) ----
  float dist = 1e10f;
  int far = 0;
  for (int i = 0; i < MCTR; ++i) {
    if (lane == 0) { fps[b * MCTR + i] = far; sidx[i] = far; }
    float cx = __shfl(x, far, 32);
    float cy = __shfl(y, far, 32);
    float cz = __shfl(z, far, 32);
    float dx = x - cx, dy = y - cy, dz = z - cz;
    float d2 = dx * dx + dy * dy + dz * dz;
    dist = fminf(dist, d2);
    // argmax reduction, ties -> lowest index (jnp.argmax semantics)
    float bd = dist; int bi = lane;
    #pragma unroll
    for (int off = 16; off > 0; off >>= 1) {
      float od = __shfl_xor(bd, off, 32);
      int   oi = __shfl_xor(bi, off, 32);
      if (od > bd || (od == bd && oi < bi)) { bd = od; bi = oi; }
    }
    far = bi;
  }
  __syncthreads();

  // ---- 2-NN for each sampled center (self included; ties -> lowest idx) ----
  for (int i = 0; i < MCTR; ++i) {
    int s = sidx[i];
    float cx = __shfl(x, s, 32);
    float cy = __shfl(y, s, 32);
    float cz = __shfl(z, s, 32);
    float dx = x - cx, dy = y - cy, dz = z - cz;
    float d2 = dx * dx + dy * dy + dz * dz;

    float bd = d2; int bi = lane;
    #pragma unroll
    for (int off = 16; off > 0; off >>= 1) {
      float od = __shfl_xor(bd, off, 32);
      int   oi = __shfl_xor(bi, off, 32);
      if (od < bd || (od == bd && oi < bi)) { bd = od; bi = oi; }
    }
    int n0 = bi;
    float d2b = (lane == n0) ? 3.4e38f : d2;
    bd = d2b; bi = lane;
    #pragma unroll
    for (int off = 16; off > 0; off >>= 1) {
      float od = __shfl_xor(bd, off, 32);
      int   oi = __shfl_xor(bi, off, 32);
      if (od < bd || (od == bd && oi < bi)) { bd = od; bi = oi; }
    }
    if (lane == 0) {
      knn[b * (MCTR * KNN_K) + i * KNN_K + 0] = n0;
      knn[b * (MCTR * KNN_K) + i * KNN_K + 1] = bi;
    }
  }

  // ---- coor_s output [b, m, 3] ----
  if (lane < MCTR) {
    int s = sidx[lane];
    coor_out[(b * MCTR + lane) * 3 + 0] = cb[s * 3 + 0];
    coor_out[(b * MCTR + lane) * 3 + 1] = cb[s * 3 + 1];
    coor_out[(b * MCTR + lane) * 3 + 2] = cb[s * 3 + 2];
  }
}

// --------------------------- X-matrix gather loaders -----------------------
struct SpLoader {  // X[1024, 4096], col = b*32 + m*2 + kk
  const float* fea; const int* fps; const int* knn;
  __device__ float loadX(int c, int col) const {
    int b = col >> 5, r = col & 31, m = r >> 1, kk = r & 1;
    int s = fps[b * MCTR + m];
    int j = knn[b * (MCTR * KNN_K) + m * KNN_K + kk];
    const float* base = fea + (size_t)b * SP_IN * NSTK;
    if (c < SP_IN) return base[c * NSTK + j] - base[c * NSTK + s];
    return base[(c - SP_IN) * NSTK + s];
  }
};

struct DnLoader {  // X[512, 131072], col = ((b*16+m)*32+p)*2 + n
  const float* fea; const int* fps; const int* knn;
  __device__ float loadX(int c, int col) const {
    int n = col & 1, p = (col >> 1) & 31, bm = col >> 6;
    int m = bm & 15, b = bm >> 4;
    int s = fps[b * MCTR + m];
    int j = knn[b * (MCTR * KNN_K) + m * KNN_K + n];
    const float* base = fea + (size_t)b * DN_IN * NSTK * NPNT;
    if (c < DN_IN) return base[(c * NSTK + j) * NPNT + p] - base[(c * NSTK + s) * NPNT + p];
    return base[((c - DN_IN) * NSTK + s) * NPNT + p];
  }
};

struct DsLoader {  // im2col: X[768, 32768], c = ci*3+t, col = (b*16+m)*16+q
  const float* h;  // [b,256,16,32]
  __device__ float loadX(int c, int col) const {
    int t = c % 3, ci = c / 3;
    int q = col & 15, bm = col >> 4, m = bm & 15, b = bm >> 4;
    int p = 2 * q - 1 + t;                     // stride 2, pad 1, kernel 3
    if (p < 0 || p >= NPNT) return 0.f;
    return h[(((size_t)b * DN_OUT + ci) * MCTR + m) * NPNT + p];
  }
};

// ------------------------------- WMMA GEMM ---------------------------------
static __device__ inline v16bf make16(v8bf lo, v8bf hi) {
  v16bf r;
  #pragma unroll
  for (int i = 0; i < 8; ++i) { r[i] = lo[i]; r[8 + i] = hi[i]; }
  return r;
}

// C[M,N] = W[M,K] * X[K,N], row-major W, X gathered on the fly.
// Block tile 128x128, 8 waves (4 along M x 2 along N), each wave 2x4 WMMA
// tiles of 16x16, K stepped by 32 (bf16). LDS tiles stored [row][K] with an
// 80-byte stride: fragments come out as two 16B ds_load_b128 per matrix,
// matching the CDNA5 16-bit A 16x32 / B 32x16 VGPR layouts.
template <class Loader>
__global__ __launch_bounds__(256) void gemm_bf16_wmma(
    const float* __restrict__ W, float* __restrict__ Y,
    int N, int K, Loader ldr)
{
  constexpr int BM = 128, BN = 128, KS = 32, LS = 40;  // LS in bf16 elems (80B)
  __shared__ __align__(16) __bf16 As[BM * LS];
  __shared__ __align__(16) __bf16 Bs[BN * LS];

  const int tid  = threadIdx.x;
  const int lane = tid & 31;
  const int wid  = tid >> 5;
  const int bm0  = blockIdx.y * BM;
  const int bn0  = blockIdx.x * BN;
  const int wm   = (wid >> 1) * 32;
  const int wn   = (wid & 1) * 64;
  const int half = lane >> 4;
  const int l15  = lane & 15;

  v8f acc[2][4];
  const v8f vz = {0.f, 0.f, 0.f, 0.f, 0.f, 0.f, 0.f, 0.f};
  #pragma unroll
  for (int mt = 0; mt < 2; ++mt)
    #pragma unroll
    for (int nt = 0; nt < 4; ++nt) acc[mt][nt] = vz;

  for (int k0 = 0; k0 < K; k0 += KS) {
    // cooperative load + f32->bf16 convert: 128 rows x 32 k each for A and B
    #pragma unroll
    for (int i = 0; i < 16; ++i) {
      int r  = i * 8 + (tid >> 5);
      int kc = tid & 31;
      As[r * LS + kc] = (__bf16)W[(size_t)(bm0 + r) * K + (k0 + kc)];
      Bs[r * LS + kc] = (__bf16)ldr.loadX(k0 + kc, bn0 + r);
    }
    __syncthreads();

    // A fragment: lane L -> M = l15, K in {kb..kb+7, 16+kb..16+kb+7}, kb=8*half
    v16bf af[2];
    #pragma unroll
    for (int mt = 0; mt < 2; ++mt) {
      int row = wm + mt * 16 + l15;
      v8bf lo = *reinterpret_cast<const v8bf*>(&As[row * LS + half * 8]);
      v8bf hi = *reinterpret_cast<const v8bf*>(&As[row * LS + 16 + half * 8]);
      af[mt] = make16(lo, hi);
    }
    // B fragment: lane L -> N = l15, K in {kb..kb+7, 16+kb..16+kb+7}, kb=8*half
    v16bf bfr[4];
    #pragma unroll
    for (int nt = 0; nt < 4; ++nt) {
      int row = wn + nt * 16 + l15;
      v8bf lo = *reinterpret_cast<const v8bf*>(&Bs[row * LS + half * 8]);
      v8bf hi = *reinterpret_cast<const v8bf*>(&Bs[row * LS + 16 + half * 8]);
      bfr[nt] = make16(lo, hi);
    }

    #pragma unroll
    for (int mt = 0; mt < 2; ++mt)
      #pragma unroll
      for (int nt = 0; nt < 4; ++nt)
        acc[mt][nt] = __builtin_amdgcn_wmma_f32_16x16x32_bf16(
            false, af[mt], false, bfr[nt], (short)0, acc[mt][nt], false, false);
    __syncthreads();
  }

  // C/D layout: lane L, vgpr v -> M = v + 8*half, N = l15
  #pragma unroll
  for (int mt = 0; mt < 2; ++mt)
    #pragma unroll
    for (int nt = 0; nt < 4; ++nt)
      #pragma unroll
      for (int v = 0; v < 8; ++v) {
        int row = bm0 + wm + mt * 16 + half * 8 + v;
        int col = bn0 + wn + nt * 16 + l15;
        Y[(size_t)row * N + col] = acc[mt][nt][v];
      }
}

// --------------------------- per-channel stats -----------------------------
__global__ __launch_bounds__(256) void channel_stats_kernel(
    const float* __restrict__ Y, int N,
    float* __restrict__ mean, float* __restrict__ var)
{
  __shared__ float ss[256];
  __shared__ float sq[256];
  const int ch = blockIdx.x, tid = threadIdx.x;
  const float* row = Y + (size_t)ch * N;
  float s = 0.f, q = 0.f;
  for (int i = tid; i < N; i += 256) { float v = row[i]; s += v; q = fmaf(v, v, q); }
  ss[tid] = s; sq[tid] = q;
  __syncthreads();
  for (int o = 128; o > 0; o >>= 1) {
    if (tid < o) { ss[tid] += ss[tid + o]; sq[tid] += sq[tid + o]; }
    __syncthreads();
  }
  if (tid == 0) {
    float mu = ss[0] / (float)N;
    mean[ch] = mu;
    var[ch] = sq[0] / (float)N - mu * mu;
  }
}

static __device__ inline float lrelu(float t) { return t > 0.f ? t : SLOPE * t; }

// -------- BN + LeakyReLU + max over k=2 (sparse) -> out [b,512,16] ---------
__global__ __launch_bounds__(256) void sp_final_kernel(
    const float* __restrict__ Y, const float* __restrict__ mean,
    const float* __restrict__ var, const float* __restrict__ g,
    const float* __restrict__ be, float* __restrict__ out)
{
  int t = blockIdx.x * 256 + threadIdx.x;          // [b][o][m]
  int m = t & 15, o = (t >> 4) & (SP_OUT - 1), b = t >> 13;
  float a = g[o] * rsqrtf(var[o] + EPSBN);
  float c = be[o] - mean[o] * a;
  size_t col = (size_t)b * 32 + m * 2;
  float y0 = lrelu(Y[(size_t)o * 4096 + col] * a + c);
  float y1 = lrelu(Y[(size_t)o * 4096 + col + 1] * a + c);
  out[t] = fmaxf(y0, y1);
}

// -------- BN + act + max over k=2 (dense) -> H [b,256,16,32] ---------------
__global__ __launch_bounds__(256) void dn_h_kernel(
    const float* __restrict__ Y, const float* __restrict__ mean,
    const float* __restrict__ var, const float* __restrict__ g,
    const float* __restrict__ be, float* __restrict__ H)
{
  int t = blockIdx.x * 256 + threadIdx.x;          // [b][o][m][p]
  int p = t & 31, m = (t >> 5) & 15, o = (t >> 9) & (DN_OUT - 1), b = t >> 17;
  float a = g[o] * rsqrtf(var[o] + EPSBN);
  float c = be[o] - mean[o] * a;
  size_t col = (((size_t)b * MCTR + m) * NPNT + p) * 2;
  float y0 = lrelu(Y[(size_t)o * 131072 + col] * a + c);
  float y1 = lrelu(Y[(size_t)o * 131072 + col + 1] * a + c);
  H[t] = fmaxf(y0, y1);
}

// -------- BN + act (downsample conv) -> out [b,256,16,16] ------------------
__global__ __launch_bounds__(256) void ds_final_kernel(
    const float* __restrict__ Y, const float* __restrict__ mean,
    const float* __restrict__ var, const float* __restrict__ g,
    const float* __restrict__ be, float* __restrict__ out)
{
  int t = blockIdx.x * 256 + threadIdx.x;          // [b][o][m][q]
  int q = t & 15, m = (t >> 4) & 15, o = (t >> 8) & (DN_OUT - 1), b = t >> 16;
  float a = g[o] * rsqrtf(var[o] + EPSBN);
  float c = be[o] - mean[o] * a;
  size_t col = ((size_t)b * MCTR + m) * 16 + q;
  out[t] = lrelu(Y[(size_t)o * 32768 + col] * a + c);
}

// ------------------------------- launcher ----------------------------------
extern "C" void kernel_launch(void* const* d_in, const int* in_sizes, int n_in,
                              void* d_out, int out_size, void* d_ws, size_t ws_size,
                              hipStream_t stream) {
  (void)in_sizes; (void)n_in; (void)out_size; (void)ws_size;

  const float* sparse_fea = (const float*)d_in[0];
  const float* dense_fea  = (const float*)d_in[1];
  const float* stk_coor   = (const float*)d_in[2];
  const float* sp_w  = (const float*)d_in[3];
  const float* sp_g  = (const float*)d_in[5];
  const float* sp_be = (const float*)d_in[6];
  const float* dn_w  = (const float*)d_in[7];
  const float* dn_g  = (const float*)d_in[9];
  const float* dn_be = (const float*)d_in[10];
  const float* ds_w  = (const float*)d_in[11];
  const float* ds_g  = (const float*)d_in[13];
  const float* ds_be = (const float*)d_in[14];
  // biases (d_in[4,8,12]) cancel inside batch-stat BN; n_stk_center fixed at 16.

  // ---- workspace carve (~243 MB) ----
  char* wsb = (char*)d_ws;
  int*   fps     = (int*)(wsb + 0);                         // 128*16
  int*   knn     = (int*)(wsb + 8192);                      // 128*32
  float* sp_mean = (float*)(wsb + 24576);
  float* sp_var  = sp_mean + SP_OUT;
  float* dn_mean = sp_var + SP_OUT;
  float* dn_var  = dn_mean + DN_OUT;
  float* ds_mean = dn_var + DN_OUT;
  float* ds_var  = ds_mean + DN_OUT;
  float* Ysp = (float*)(wsb + 32768);                       // 512 x 4096
  float* Ydn = Ysp + (size_t)SP_OUT * 4096;                 // 256 x 131072
  float* H   = Ydn + (size_t)DN_OUT * 131072;               // 128*256*16*32
  float* Yds = H + (size_t)BS * DN_OUT * MCTR * NPNT;       // 256 x 32768

  float* out      = (float*)d_out;
  float* out_sp   = out;                                    // 1,048,576
  float* out_dn   = out + (size_t)BS * SP_OUT * MCTR;       // 8,388,608
  float* out_coor = out_dn + (size_t)BS * DN_OUT * MCTR * (NPNT / 2);

  // 1) FPS + KNN + coor_s
  fps_knn_kernel<<<BS, 32, 0, stream>>>(stk_coor, fps, knn, out_coor);

  // 2) sparse GEMM: [512,4096] = sp_w[512,1024] x X_sp
  SpLoader spl{sparse_fea, fps, knn};
  gemm_bf16_wmma<<<dim3(4096 / 128, SP_OUT / 128), 256, 0, stream>>>(
      sp_w, Ysp, 4096, 2 * SP_IN, spl);
  channel_stats_kernel<<<SP_OUT, 256, 0, stream>>>(Ysp, 4096, sp_mean, sp_var);
  sp_final_kernel<<<(BS * SP_OUT * MCTR) / 256, 256, 0, stream>>>(
      Ysp, sp_mean, sp_var, sp_g, sp_be, out_sp);

  // 3) dense GEMM: [256,131072] = dn_w[256,512] x X_dn
  DnLoader dnl{dense_fea, fps, knn};
  gemm_bf16_wmma<<<dim3(131072 / 128, DN_OUT / 128), 256, 0, stream>>>(
      dn_w, Ydn, 131072, 2 * DN_IN, dnl);
  channel_stats_kernel<<<DN_OUT, 256, 0, stream>>>(Ydn, 131072, dn_mean, dn_var);
  dn_h_kernel<<<(BS * DN_OUT * MCTR * NPNT) / 256, 256, 0, stream>>>(
      Ydn, dn_mean, dn_var, dn_g, dn_be, H);

  // 4) downsample conv as im2col GEMM: [256,32768] = ds_w[256,768] x X_ds
  DsLoader dsl{H};
  gemm_bf16_wmma<<<dim3(32768 / 128, DN_OUT / 128), 256, 0, stream>>>(
      ds_w, Yds, 32768, DN_IN * 3, dsl);
  channel_stats_kernel<<<DN_OUT, 256, 0, stream>>>(Yds, 32768, ds_mean, ds_var);
  ds_final_kernel<<<(BS * DN_OUT * MCTR * (NPNT / 2)) / 256, 256, 0, stream>>>(
      Yds, ds_mean, ds_var, ds_g, ds_be, out_dn);
}
